// EntropyGeoRouter_71940702208340
// MI455X (gfx1250) — matrix operand order repaired and checked by reading, not compiled, not run on hardware
//
#include <hip/hip_runtime.h>
#include <math.h>

typedef __attribute__((ext_vector_type(16))) __bf16 v16bf;
typedef __attribute__((ext_vector_type(8)))  float  v8f;

constexpr int Bn = 4, Sn = 1024, Dn = 256, NSn = 64, KNBn = 8;
constexpr float NEGV = -10000.0f;
constexpr float LOGV = 10.373491181781864f;   // log(32000)

// ---------- bf16 helpers (storage as u16, RNE convert) ----------
__device__ inline unsigned short f2bf(float f) {
    unsigned u = __float_as_uint(f);
    return (unsigned short)((u + 0x7FFFu + ((u >> 16) & 1u)) >> 16);
}
__device__ inline float bf2f(unsigned short h) {
    return __uint_as_float(((unsigned)h) << 16);
}

// ---------- WMMA fragments (wave32, 16x16x32 bf16) ----------
struct Frag { v16bf v; };

// A-matrix 16x32: lane L holds row m=L&15; K runs [hi*8, hi*8+8) and [16+hi*8, ...)
__device__ inline Frag load_a(const unsigned short* base, int stride, int kbase, int lane) {
    Frag f; uint4* q = reinterpret_cast<uint4*>(&f);
    int m = lane & 15, hi = (lane >> 4) & 1;
    const unsigned short* p = base + (size_t)m * stride + kbase + hi * 8;
    q[0] = *reinterpret_cast<const uint4*>(p);
    q[1] = *reinterpret_cast<const uint4*>(p + 16);
    return f;
}
// B-matrix 32x16 with element (k,n) at base[(nbase+n)*stride + kbase + k]:
// lane L holds n=L&15; K runs [hi*8, hi*8+8) and [16+hi*8, ...) -> two b128 loads
__device__ inline Frag load_bt(const unsigned short* base, int stride, int nbase, int kbase, int lane) {
    Frag f; uint4* q = reinterpret_cast<uint4*>(&f);
    const unsigned short* p = base + (size_t)(nbase + (lane & 15)) * stride + kbase + ((lane >> 4) & 1) * 8;
    q[0] = *reinterpret_cast<const uint4*>(p);
    q[1] = *reinterpret_cast<const uint4*>(p + 16);
    return f;
}
__device__ inline v8f wmma_bf(Frag a, Frag b, v8f c) {
    return __builtin_amdgcn_wmma_f32_16x16x32_bf16(false, a.v, false, b.v, (short)0, c, false, false);
}

// ---------- block reduction (256 threads) ----------
__device__ inline float bsum(float v, float* red) {
    int t = threadIdx.x;
    red[t] = v; __syncthreads();
    for (int o = 128; o > 0; o >>= 1) { if (t < o) red[t] += red[t + o]; __syncthreads(); }
    float r = red[0]; __syncthreads();
    return r;
}

// ================= K1: prep (normalize, bf16 staging, gates, modes) =================
__global__ void __launch_bounds__(256) k_prep(
    const float* __restrict__ msg, const float* __restrict__ hid,
    const float* __restrict__ scn, const float* __restrict__ gv,
    const float* __restrict__ ent, const float* __restrict__ ctx,
    const float* __restrict__ Wvel, const float* __restrict__ Wgate, const float* __restrict__ bgate,
    unsigned short* msgb, unsigned short* msgT, unsigned short* haimb, unsigned short* hidb,
    unsigned short* scnb, unsigned short* epb, int* mode, float* Hn, float* gw)
{
    int s = blockIdx.x, b = blockIdx.y, t = threadIdx.x;
    __shared__ float red[256];
    __shared__ float gvv[NSn], scv[NSn], epr[NSn];
    size_t rowNS = ((size_t)b * Sn + s) * NSn;
    size_t rowD  = ((size_t)b * Sn + s) * Dn;

    if (t < NSn) {
        float g = gv[rowNS + t], sc = scn[rowNS + t];
        gvv[t] = g; scv[t] = sc; epr[t] = sc + 0.4f * g;
    }
    __syncthreads();
    float nrm2 = bsum((t < NSn) ? epr[t] * epr[t] : 0.f, red);
    float nrm = fmaxf(sqrtf(nrm2), 1e-12f);
    if (t < NSn) {
        epb[rowNS + t]  = f2bf(epr[t] / nrm);
        scnb[rowNS + t] = f2bf(scv[t]);
    }
    // vel_h = gv @ Wvel^T ; h_aimed = normalize(hidden + 0.3*vel_h)
    float hv = hid[rowD + t];
    float acc = 0.f;
    #pragma unroll 8
    for (int n = 0; n < NSn; ++n) acc += gvv[n] * Wvel[(size_t)t * NSn + n];
    float h = hv + 0.3f * acc;
    float hn2 = bsum(h * h, red);
    float hnm = fmaxf(sqrtf(hn2), 1e-12f);
    haimb[rowD + t] = f2bf(h / hnm);
    hidb[rowD + t]  = f2bf(hv);
    float m = msg[rowD + t];
    msgb[rowD + t] = f2bf(m);
    msgT[((size_t)b * Dn + t) * Sn + s] = f2bf(m);
    // gate / entropy / mode
    float gdot = bsum(hv * Wgate[t], red);
    if (t == 0) {
        float g = 1.f / (1.f + expf(-(gdot + bgate[0])));
        gw[(size_t)b * Sn + s] = g * ctx[(size_t)b * Sn + s];
        Hn[(size_t)b * Sn + s] = ent[(size_t)b * Sn + s] / (LOGV + 1e-8f);
        int bm = 0; float bv = scv[0];
        for (int n = 1; n < NSn; ++n) if (scv[n] > bv) { bv = scv[n]; bm = n; }
        mode[(size_t)b * Sn + s] = bm;
    }
}

// ================= tiny: f32 -> bf16 weight conversion =================
__global__ void __launch_bounds__(256) k_convW(const float* __restrict__ W,
                                               unsigned short* __restrict__ Wb, int n)
{
    int i = blockIdx.x * 256 + threadIdx.x;
    if (i < n) Wb[i] = f2bf(W[i]);
}

// ================= K2: geo branch (WMMA scores -> sparsemax -> WMMA aggregate) =====
__global__ void __launch_bounds__(128) k_geo(
    const unsigned short* __restrict__ haimb, const unsigned short* __restrict__ epb,
    const unsigned short* __restrict__ msgb, const unsigned short* __restrict__ scnb,
    const unsigned short* __restrict__ msgT, const float* __restrict__ Hn,
    const unsigned char* __restrict__ mask, float* __restrict__ geo)
{
    int b = blockIdx.y, row = blockIdx.x * 16;
    int tid = threadIdx.x, wave = tid >> 5, lane = tid & 31;
    int hi = (lane >> 4) & 1, nlo = lane & 15;
    __shared__ __align__(16) unsigned short sc[16 * Sn];   // 32 KB bf16 score strip

    // A fragments kept in registers, reused across all 64 column tiles
    Frag ah[8];
    const unsigned short* abase = haimb + ((size_t)b * Sn + row) * Dn;
    #pragma unroll
    for (int kc = 0; kc < 8; ++kc) ah[kc] = load_a(abase, Dn, kc * 32, lane);
    Frag ae[2];
    const unsigned short* ebase = epb + ((size_t)b * Sn + row) * NSn;
    ae[0] = load_a(ebase, NSn, 0, lane);
    ae[1] = load_a(ebase, NSn, 32, lane);
    float hnv[8];
    #pragma unroll
    for (int v = 0; v < 8; ++v) hnv[v] = Hn[(size_t)b * Sn + row + v + hi * 8];

    const unsigned short* mrow = msgb + (size_t)b * Sn * Dn;
    const unsigned short* srow = scnb + (size_t)b * Sn * NSn;
    for (int ct = wave; ct < Sn / 16; ct += 4) {
        int col = ct * 16;
        // prefetch next tile's B stream one iteration ahead
        if (ct + 4 < Sn / 16)
            __builtin_prefetch(mrow + (size_t)((ct + 4) * 16 + nlo) * Dn + hi * 128, 0, 1);
        v8f ch = {0,0,0,0,0,0,0,0}, ce = {0,0,0,0,0,0,0,0};
        #pragma unroll
        for (int kc = 0; kc < 8; ++kc) { Frag bb = load_bt(mrow, Dn, col, kc * 32, lane); ch = wmma_bf(ah[kc], bb, ch); }
        #pragma unroll
        for (int kc = 0; kc < 2; ++kc) { Frag bb = load_bt(srow, NSn, col, kc * 32, lane); ce = wmma_bf(ae[kc], bb, ce); }
        int tcol = col + nlo;
        #pragma unroll
        for (int v = 0; v < 8; ++v) {
            int si = row + v + hi * 8;
            float comb = (ce[v] * hnv[v] + 0.5f * ch[v]) * 5.f;
            if (mask[(size_t)si * Sn + tcol] || si == tcol) comb = NEGV;
            sc[(v + hi * 8) * Sn + tcol] = f2bf(comb);
        }
    }
    __syncthreads();

    // sparsemax via binary search on tau (monotone), 4 rows per wave
    for (int i = 0; i < 4; ++i) {
        int r = wave * 4 + i;
        float z[32];
        #pragma unroll
        for (int j = 0; j < 32; ++j) z[j] = bf2f(sc[r * Sn + lane + 32 * j]);
        float mx = z[0];
        #pragma unroll
        for (int j = 1; j < 32; ++j) mx = fmaxf(mx, z[j]);
        for (int o = 16; o; o >>= 1) mx = fmaxf(mx, __shfl_xor(mx, o));
        float lo = mx - 1.f, hi2 = mx;
        for (int it = 0; it < 30; ++it) {
            float mid = 0.5f * (lo + hi2), sm = 0.f;
            #pragma unroll
            for (int j = 0; j < 32; ++j) sm += fmaxf(z[j] - mid, 0.f);
            for (int o = 16; o; o >>= 1) sm += __shfl_xor(sm, o);
            if (sm > 1.f) lo = mid; else hi2 = mid;
        }
        float tau = 0.5f * (lo + hi2);
        #pragma unroll
        for (int j = 0; j < 32; ++j) sc[r * Sn + lane + 32 * j] = f2bf(fmaxf(z[j] - tau, 0.f));
    }
    __syncthreads();

    // geo_agg = ew @ messages  (A from LDS, B from msgT [D][S] -> contiguous K)
    const unsigned short* tbase = msgT + (size_t)b * Dn * Sn;
    int dbase = wave * 64;
    v8f cc[4];
    #pragma unroll
    for (int tI = 0; tI < 4; ++tI) cc[tI] = (v8f){0,0,0,0,0,0,0,0};
    for (int kc = 0; kc < Sn / 32; ++kc) {
        if (kc + 1 < Sn / 32)
            __builtin_prefetch(tbase + (size_t)(dbase + nlo) * Sn + (kc + 1) * 32 + hi * 16, 0, 1);
        Frag a = load_a(sc, Sn, kc * 32, lane);
        #pragma unroll
        for (int tI = 0; tI < 4; ++tI) {
            Frag bb = load_bt(tbase, Sn, dbase + tI * 16, kc * 32, lane);
            cc[tI] = wmma_bf(a, bb, cc[tI]);
        }
    }
    #pragma unroll
    for (int tI = 0; tI < 4; ++tI)
        #pragma unroll
        for (int v = 0; v < 8; ++v)
            geo[((size_t)b * Sn + row + v + hi * 8) * Dn + dbase + tI * 16 + nlo] = cc[tI][v];
}

// ================= K3: local-mode branch =================
__global__ void __launch_bounds__(128) k_local(
    const unsigned short* __restrict__ scnb, const unsigned short* __restrict__ msgT,
    const int* __restrict__ mode, const unsigned char* __restrict__ mask,
    float* __restrict__ loc)
{
    int b = blockIdx.y, row = blockIdx.x * 16;
    int tid = threadIdx.x, wave = tid >> 5, lane = tid & 31;
    int hi = (lane >> 4) & 1, nlo = lane & 15;
    __shared__ __align__(16) unsigned short sc[16 * Sn];
    __shared__ int has[16];
    if (tid < 16) has[tid] = 0;
    __syncthreads();

    Frag as2[2];
    const unsigned short* abase = scnb + ((size_t)b * Sn + row) * NSn;
    as2[0] = load_a(abase, NSn, 0, lane);
    as2[1] = load_a(abase, NSn, 32, lane);
    int modev[8];
    #pragma unroll
    for (int v = 0; v < 8; ++v) modev[v] = mode[(size_t)b * Sn + row + v + hi * 8];

    const unsigned short* srow = scnb + (size_t)b * Sn * NSn;
    for (int ct = wave; ct < Sn / 16; ct += 4) {
        int col = ct * 16;
        v8f ce = {0,0,0,0,0,0,0,0};
        #pragma unroll
        for (int kc = 0; kc < 2; ++kc) { Frag bb = load_bt(srow, NSn, col, kc * 32, lane); ce = wmma_bf(as2[kc], bb, ce); }
        int tcol = col + nlo;
        int mt = mode[(size_t)b * Sn + tcol];
        #pragma unroll
        for (int v = 0; v < 8; ++v) {
            int si = row + v + hi * 8;
            bool blk = mask[(size_t)si * Sn + tcol] || (si == tcol);
            bool same = (modev[v] == mt);
            float val = blk ? NEGV : (same ? ce[v] : 0.f);
            if (!blk && same) atomicOr(&has[v + hi * 8], 1);
            sc[(v + hi * 8) * Sn + tcol] = f2bf(val);
        }
    }
    __syncthreads();

    for (int i = 0; i < 4; ++i) {
        int r = wave * 4 + i;
        int si = row + r;
        int hb = has[r];
        float z[32];
        #pragma unroll
        for (int j = 0; j < 32; ++j) {
            int tcol = lane + 32 * j;
            if (hb) z[j] = 5.f * bf2f(sc[r * Sn + tcol]);
            else {
                bool blk = mask[(size_t)si * Sn + tcol] || (si == tcol);
                z[j] = blk ? 5.f * NEGV : -fabsf((float)(si - tcol)) * 0.05f;   // pos*0.01*5
            }
        }
        float mx = z[0];
        #pragma unroll
        for (int j = 1; j < 32; ++j) mx = fmaxf(mx, z[j]);
        for (int o = 16; o; o >>= 1) mx = fmaxf(mx, __shfl_xor(mx, o));
        float lo = mx - 1.f, hi2 = mx;
        for (int it = 0; it < 30; ++it) {
            float mid = 0.5f * (lo + hi2), sm = 0.f;
            #pragma unroll
            for (int j = 0; j < 32; ++j) sm += fmaxf(z[j] - mid, 0.f);
            for (int o = 16; o; o >>= 1) sm += __shfl_xor(sm, o);
            if (sm > 1.f) lo = mid; else hi2 = mid;
        }
        float tau = 0.5f * (lo + hi2);
        #pragma unroll
        for (int j = 0; j < 32; ++j) sc[r * Sn + lane + 32 * j] = f2bf(fmaxf(z[j] - tau, 0.f));
    }
    __syncthreads();

    const unsigned short* tbase = msgT + (size_t)b * Dn * Sn;
    int dbase = wave * 64;
    v8f cc[4];
    #pragma unroll
    for (int tI = 0; tI < 4; ++tI) cc[tI] = (v8f){0,0,0,0,0,0,0,0};
    for (int kc = 0; kc < Sn / 32; ++kc) {
        if (kc + 1 < Sn / 32)
            __builtin_prefetch(tbase + (size_t)(dbase + nlo) * Sn + (kc + 1) * 32 + hi * 16, 0, 1);
        Frag a = load_a(sc, Sn, kc * 32, lane);
        #pragma unroll
        for (int tI = 0; tI < 4; ++tI) {
            Frag bb = load_bt(tbase, Sn, dbase + tI * 16, kc * 32, lane);
            cc[tI] = wmma_bf(a, bb, cc[tI]);
        }
    }
    #pragma unroll
    for (int tI = 0; tI < 4; ++tI)
        #pragma unroll
        for (int v = 0; v < 8; ++v)
            loc[((size_t)b * Sn + row + v + hi * 8) * Dn + dbase + tI * 16 + nlo] = cc[tI][v];
}

// ================= K-probe: probe = hidden @ W_probe^T as tiled WMMA GEMM =========
__global__ void __launch_bounds__(128) k_probe(
    const unsigned short* __restrict__ hidb, const unsigned short* __restrict__ Wpb,
    float* __restrict__ probe)
{
    int b = blockIdx.y, row = blockIdx.x * 16;
    int tid = threadIdx.x, wave = tid >> 5, lane = tid & 31;
    int hi = (lane >> 4) & 1, nlo = lane & 15;
    Frag ah[8];
    const unsigned short* abase = hidb + ((size_t)b * Sn + row) * Dn;
    #pragma unroll
    for (int kc = 0; kc < 8; ++kc) ah[kc] = load_a(abase, Dn, kc * 32, lane);
    int dbase = wave * 64;
    v8f cc[4];
    #pragma unroll
    for (int tI = 0; tI < 4; ++tI) cc[tI] = (v8f){0,0,0,0,0,0,0,0};
    #pragma unroll
    for (int kc = 0; kc < 8; ++kc) {
        #pragma unroll
        for (int tI = 0; tI < 4; ++tI) {
            // probe[s][d] = sum_i hid[s][i] * Wp[d][i]  -> B[k][n] = Wp[dbase+n][k]
            Frag bb = load_bt(Wpb, Dn, dbase + tI * 16, kc * 32, lane);
            cc[tI] = wmma_bf(ah[kc], bb, cc[tI]);
        }
    }
    #pragma unroll
    for (int tI = 0; tI < 4; ++tI)
        #pragma unroll
        for (int v = 0; v < 8; ++v)
            probe[((size_t)b * Sn + row + v + hi * 8) * Dn + dbase + tI * 16 + nlo] = cc[tI][v];
}

// ================= K4: static neighbor-vocab branch (sparse scan) =================
__global__ void __launch_bounds__(256) k_static(
    const float* __restrict__ msg, const float* __restrict__ scn,
    const int* __restrict__ xids, const int* __restrict__ nbtab,
    const float* __restrict__ We1, const float* __restrict__ be1,
    const float* __restrict__ We2, const float* __restrict__ be2,
    float* __restrict__ statagg, float* __restrict__ cov)
{
    int s = blockIdx.x, b = blockIdx.y, t = threadIdx.x;
    __shared__ float macc[KNBn][Dn];
    __shared__ float sacc[KNBn][NSn];
    __shared__ int   cnt[KNBn];
    __shared__ int   nbid[KNBn];
    __shared__ float scv[NSn];
    __shared__ float ew[KNBn];
    __shared__ float simv[KNBn];
    for (int i = t; i < KNBn * Dn;  i += 256) ((float*)macc)[i] = 0.f;
    for (int i = t; i < KNBn * NSn; i += 256) ((float*)sacc)[i] = 0.f;
    if (t < KNBn) { cnt[t] = 0; nbid[t] = nbtab[(size_t)xids[(size_t)b * Sn + s] * KNBn + t]; }
    if (t < NSn) scv[t] = scn[((size_t)b * Sn + s) * NSn + t];
    __syncthreads();

    for (int p = t; p <= s; p += 256) {                 // causal p <= s
        int x = xids[(size_t)b * Sn + p];
        #pragma unroll
        for (int k = 0; k < KNBn; ++k) {
            if (x == nbid[k]) {                         // expected rate ~ s/32000
                atomicAdd(&cnt[k], 1);
                const float* mr = msg + ((size_t)b * Sn + p) * Dn;
                for (int d = 0; d < Dn; ++d) atomicAdd(&macc[k][d], mr[d]);
                const float* sr = scn + ((size_t)b * Sn + p) * NSn;
                for (int n = 0; n < NSn; ++n) atomicAdd(&sacc[k][n], sr[n]);
            }
        }
    }
    __syncthreads();
    if (t < KNBn) {
        float a = 0.f;
        for (int n = 0; n < NSn; ++n) a += sacc[t][n] * scv[n];
        simv[t] = a;
    }
    __syncthreads();
    if (t == 0) {
        float e[KNBn]; float mx = -1e30f;
        for (int k = 0; k < KNBn; ++k) {
            float x = simv[k], o = 0.f;
            for (int j = 0; j < 8; ++j) {
                float pre = x * We1[j] + be1[j];
                float g = 0.5f * pre * (1.f + erff(pre * 0.70710678f));  // exact gelu
                o += We2[j] * g;
            }
            e[k] = o + be2[0]; mx = fmaxf(mx, e[k]);
        }
        float ssum = 0.f;
        for (int k = 0; k < KNBn; ++k) { ew[k] = expf(e[k] - mx); ssum += ew[k]; }
        for (int k = 0; k < KNBn; ++k) ew[k] /= ssum;
        float c = 0.f;
        for (int k = 0; k < KNBn; ++k) c += fminf((float)cnt[k], 1.f);
        cov[(size_t)b * Sn + s] = c / 8.f;
    }
    __syncthreads();
    float a = 0.f;
    #pragma unroll
    for (int k = 0; k < KNBn; ++k) a += macc[k][t] * ew[k];
    statagg[((size_t)b * Sn + s) * Dn + t] = a;
}

// ================= K5: gated fusion + relevance (probe precomputed) =================
__global__ void __launch_bounds__(256) k_fuse(
    const float* __restrict__ probe,
    const float* __restrict__ gw, const float* __restrict__ cov,
    const float* __restrict__ geo, const float* __restrict__ loc,
    const float* __restrict__ statagg, float* __restrict__ out)
{
    int s = blockIdx.x, b = blockIdx.y, t = threadIdx.x;
    __shared__ float red[256];
    size_t rowD = ((size_t)b * Sn + s) * Dn;
    float p = probe[rowD + t];
    float pn = fmaxf(sqrtf(bsum(p * p, red)), 1e-12f);

    float g = gw[(size_t)b * Sn + s], c = cov[(size_t)b * Sn + s];
    float sf  = c * statagg[rowD + t] + (1.f - c) * loc[rowD + t];
    float agg = g * geo[rowD + t] + (1.f - g) * sf;
    float an = fmaxf(sqrtf(bsum(agg * agg, red)), 1e-12f);
    float dot = bsum((agg / an) * (p / pn), red);
    float rel = 1.f / (1.f + expf(-dot));
    out[rowD + t] = agg * rel;
}

// ================= launch =================
extern "C" void kernel_launch(void* const* d_in, const int* in_sizes, int n_in,
                              void* d_out, int out_size, void* d_ws, size_t ws_size,
                              hipStream_t stream)
{
    const float* messages = (const float*)d_in[0];
    const float* hidden   = (const float*)d_in[1];
    const int*   x_ids    = (const int*)d_in[2];
    const float* scn      = (const float*)d_in[3];
    const unsigned char* mask = (const unsigned char*)d_in[4];
    const int*   static_nb = (const int*)d_in[5];
    const float* geo_vel  = (const float*)d_in[6];
    const float* ctx_conf = (const float*)d_in[7];
    const float* entropy  = (const float*)d_in[8];
    const float* W_vel    = (const float*)d_in[9];
    const float* W_e1     = (const float*)d_in[10];
    const float* b_e1     = (const float*)d_in[11];
    const float* W_e2     = (const float*)d_in[12];
    const float* b_e2     = (const float*)d_in[13];
    const float* W_probe  = (const float*)d_in[14];
    const float* W_gate   = (const float*)d_in[15];
    const float* b_gate   = (const float*)d_in[16];
    float* out = (float*)d_out;

    char* ws = (char*)d_ws;
    size_t off = 0;
    auto alloc = [&](size_t bytes) -> void* {
        void* p = ws + off;
        off = (off + bytes + 255) & ~(size_t)255;
        return p;
    };
    unsigned short* msgb  = (unsigned short*)alloc((size_t)Bn * Sn * Dn * 2);
    unsigned short* msgT  = (unsigned short*)alloc((size_t)Bn * Dn * Sn * 2);
    unsigned short* haimb = (unsigned short*)alloc((size_t)Bn * Sn * Dn * 2);
    unsigned short* hidb  = (unsigned short*)alloc((size_t)Bn * Sn * Dn * 2);
    unsigned short* scnb  = (unsigned short*)alloc((size_t)Bn * Sn * NSn * 2);
    unsigned short* epb   = (unsigned short*)alloc((size_t)Bn * Sn * NSn * 2);
    unsigned short* Wpb   = (unsigned short*)alloc((size_t)Dn * Dn * 2);
    int*   mode  = (int*)alloc((size_t)Bn * Sn * 4);
    float* Hn    = (float*)alloc((size_t)Bn * Sn * 4);
    float* gw    = (float*)alloc((size_t)Bn * Sn * 4);
    float* cov   = (float*)alloc((size_t)Bn * Sn * 4);
    float* geo   = (float*)alloc((size_t)Bn * Sn * Dn * 4);
    float* loc   = (float*)alloc((size_t)Bn * Sn * Dn * 4);
    float* stat  = (float*)alloc((size_t)Bn * Sn * Dn * 4);
    float* probe = (float*)alloc((size_t)Bn * Sn * Dn * 4);

    dim3 gFull(Sn, Bn);
    dim3 gTile(Sn / 16, Bn);

    k_prep<<<gFull, 256, 0, stream>>>(messages, hidden, scn, geo_vel, entropy, ctx_conf,
                                      W_vel, W_gate, b_gate,
                                      msgb, msgT, haimb, hidb, scnb, epb, mode, Hn, gw);
    k_convW<<<dim3((Dn * Dn + 255) / 256), 256, 0, stream>>>(W_probe, Wpb, Dn * Dn);
    k_static<<<gFull, 256, 0, stream>>>(messages, scn, x_ids, static_nb,
                                        W_e1, b_e1, W_e2, b_e2, stat, cov);
    k_geo<<<gTile, 128, 0, stream>>>(haimb, epb, msgb, scnb, msgT, Hn, mask, geo);
    k_local<<<gTile, 128, 0, stream>>>(scnb, msgT, mode, mask, loc);
    k_probe<<<gTile, 128, 0, stream>>>(hidb, Wpb, probe);
    k_fuse<<<gFull, 256, 0, stream>>>(probe, gw, cov, geo, loc, stat, out);
}
